// GAT_node_45801531245069
// MI455X (gfx1250) — matrix operand order repaired and checked by reading, not compiled, not run on hardware
//
#include <hip/hip_runtime.h>
#include <hip/hip_bf16.h>
#include <stdint.h>

#define IN_DIM  256
#define HID     128
#define HEADS   2
#define D1      (HEADS * HID)   // 256
#define OUT_DIM 128
#define ENC_NEGINF 0x007FFFFFu  // enc(-inf)

typedef _Float16 half8  __attribute__((ext_vector_type(8)));
typedef _Float16 half16 __attribute__((ext_vector_type(16)));
typedef float    f32x8  __attribute__((ext_vector_type(8)));

#define CDIV(a, b) (((a) + (b) - 1) / (b))

// ---------------- helpers: monotone float<->uint encoding for atomicMax ----
__device__ __forceinline__ unsigned enc_f(float f) {
    unsigned u = __float_as_uint(f);
    return (u & 0x80000000u) ? ~u : (u | 0x80000000u);
}
__device__ __forceinline__ float dec_f(unsigned u) {
    u = (u & 0x80000000u) ? (u & 0x7fffffffu) : ~u;
    return __uint_as_float(u);
}

// ---------------- prep kernels --------------------------------------------
__global__ void f32_to_f16_kernel(const float* __restrict__ x,
                                  _Float16* __restrict__ y, size_t n) {
    size_t i = (size_t)blockIdx.x * blockDim.x + threadIdx.x;
    if (i < n) y[i] = (_Float16)x[i];
}

// W [K, Nout] f32 row-major  ->  Wt [Nout, K] f16 row-major (i.e. B^T)
__global__ void transpose_to_f16_kernel(const float* __restrict__ W,
                                        _Float16* __restrict__ Wt,
                                        int K, int Nout) {
    int i = blockIdx.x * blockDim.x + threadIdx.x;
    if (i >= K * Nout) return;
    int k = i / Nout, n = i - k * Nout;
    Wt[(size_t)n * K + k] = (_Float16)W[i];
}

__global__ void fill_u32_kernel(unsigned* __restrict__ p, unsigned v, size_t n) {
    size_t i = (size_t)blockIdx.x * blockDim.x + threadIdx.x;
    if (i < n) p[i] = v;
}

// ---------------- WMMA GEMM: C[M,Nout] = A[M,K](f16) x Bt[Nout,K](f16)^T ---
// One wave per 16x16 C tile, K unrolled (template), f32 accumulate.
template <int K>
__global__ void wmma_gemm_f16_kernel(const _Float16* __restrict__ A,
                                     const _Float16* __restrict__ Bt,
                                     float* __restrict__ C,
                                     int tilesM, int tilesN, int Nout) {
    const int lane = threadIdx.x & 31;
    const int gw   = blockIdx.x * (blockDim.x >> 5) + (threadIdx.x >> 5);
    if (gw >= tilesM * tilesN) return;
    const int tm = gw / tilesN;
    const int tn = gw - tm * tilesN;
    const int r  = lane & 15;
    const int hi = lane >> 4;

    // A lane layout (16x32 f16): row M = r, halves K = {hi*8..hi*8+7, 16+hi*8..16+hi*8+7}
    const _Float16* pa = A + ((size_t)(tm * 16 + r)) * K + hi * 8;
    // B lane layout (32x16 f16): col N = r, halves K = {hi*8..hi*8+7, 16+hi*8..16+hi*8+7}
    const _Float16* pb = Bt + ((size_t)(tn * 16 + r)) * K + hi * 8;

    f32x8 acc = {};
#pragma unroll
    for (int k0 = 0; k0 < K; k0 += 32) {
        union { half16 v; half8 h[2]; } ua, ub;
        ua.h[0] = *(const half8*)(pa + k0);
        ua.h[1] = *(const half8*)(pa + k0 + 16);
        ub.h[0] = *(const half8*)(pb + k0);
        ub.h[1] = *(const half8*)(pb + k0 + 16);
        if (k0 + 32 < K) {
            __builtin_prefetch(pa + k0 + 32, 0, 0);  // global_prefetch_b8
            __builtin_prefetch(pb + k0 + 32, 0, 0);
        }
        acc = __builtin_amdgcn_wmma_f32_16x16x32_f16(
            false, ua.v, false, ub.v, (short)0, acc, false, false);
    }
    // C/D layout: VGPR i -> row (hi*8 + i), col = r
    float* pc = C + ((size_t)(tm * 16 + hi * 8)) * Nout + tn * 16 + r;
#pragma unroll
    for (int i = 0; i < 8; ++i) pc[(size_t)i * Nout] = acc[i];
}

// ---------------- attention logits: a_s/a_d[n,h] = <h[n,h,:], att[h,:]> ----
__global__ void att_dots_kernel(const float* __restrict__ h,
                                const float* __restrict__ att_s,
                                const float* __restrict__ att_d,
                                float* __restrict__ a_s,
                                float* __restrict__ a_d,
                                int Ntot, int H, int C) {
    int lane = threadIdx.x & 31;
    int gw = blockIdx.x * (blockDim.x >> 5) + (threadIdx.x >> 5);
    if (gw >= Ntot * H) return;
    int n = gw / H, head = gw - n * H;
    const float* hp = h + (size_t)n * H * C + (size_t)head * C;
    const float* sp = att_s + head * C;
    const float* dp = att_d + head * C;
    float ss = 0.f, sd = 0.f;
    for (int c = lane; c < C; c += 32) {
        float v = hp[c];
        ss += v * sp[c];
        sd += v * dp[c];
    }
#pragma unroll
    for (int off = 16; off > 0; off >>= 1) {
        ss += __shfl_down(ss, off, 32);
        sd += __shfl_down(sd, off, 32);
    }
    if (lane == 0) { a_s[gw] = ss; a_d[gw] = sd; }
}

// ---------------- edge passes ---------------------------------------------
__global__ void edge_max_kernel(const int* __restrict__ src, const int* __restrict__ dst,
                                const float* __restrict__ a_s, const float* __restrict__ a_d,
                                unsigned* __restrict__ m_enc, int Etot, int H) {
    int t = blockIdx.x * blockDim.x + threadIdx.x;
    if (t >= Etot * H) return;
    int e = t / H, head = t - e * H;
    int s = src[e], d = dst[e];
    float v = a_s[s * H + head] + a_d[d * H + head];
    v = v > 0.f ? v : 0.2f * v;                 // leaky_relu(0.2)
    atomicMax(m_enc + d * H + head, enc_f(v));
}

__global__ void decode_max_kernel(unsigned* __restrict__ m, int n) {
    int i = blockIdx.x * blockDim.x + threadIdx.x;
    if (i >= n) return;
    unsigned u = m[i];
    float f = (u == ENC_NEGINF) ? 0.f : dec_f(u);  // isolated node -> 0
    ((float*)m)[i] = f;
}

__global__ void edge_w_kernel(const int* __restrict__ src, const int* __restrict__ dst,
                              const float* __restrict__ a_s, const float* __restrict__ a_d,
                              const float* __restrict__ m, float* __restrict__ den,
                              float* __restrict__ wbuf, int Etot, int H) {
    int t = blockIdx.x * blockDim.x + threadIdx.x;
    if (t >= Etot * H) return;
    int e = t / H, head = t - e * H;
    int s = src[e], d = dst[e];
    float v = a_s[s * H + head] + a_d[d * H + head];
    v = v > 0.f ? v : 0.2f * v;
    float w = __expf(v - m[d * H + head]);
    wbuf[t] = w;
    atomicAdd(den + d * H + head, w);
}

// one wave per edge: coalesced lane reads of h[src], scatter-add into agg[dst]
__global__ void edge_aggregate_kernel(const int* __restrict__ src, const int* __restrict__ dst,
                                      const float* __restrict__ h,
                                      const float* __restrict__ wbuf,
                                      const float* __restrict__ den,
                                      float* __restrict__ agg,
                                      int Etot, int H, int C) {
    int lane = threadIdx.x & 31;
    int gw = blockIdx.x * (blockDim.x >> 5) + (threadIdx.x >> 5);
    if (gw >= Etot) return;
    int s = src[gw], d = dst[gw];
    int Dtot = H * C;
    for (int dim = lane; dim < Dtot; dim += 32) {
        int head = dim / C;
        float alpha = wbuf[gw * H + head] / (den[d * H + head] + 1e-16f);
        atomicAdd(agg + (size_t)d * Dtot + dim, h[(size_t)s * Dtot + dim] * alpha);
    }
}

// ---------------- batchnorm -----------------------------------------------
__global__ void bn_colstats_kernel(const float* __restrict__ x, const float* __restrict__ bias,
                                   float* __restrict__ csum, float* __restrict__ csumsq,
                                   int Ntot, int D, int rowsPerBlock) {
    int col = threadIdx.x;                 // blockDim.x == D
    int r0 = blockIdx.x * rowsPerBlock;
    int r1 = r0 + rowsPerBlock; if (r1 > Ntot) r1 = Ntot;
    float b = bias[col];
    float s = 0.f, s2 = 0.f;
    for (int r = r0; r < r1; ++r) {
        float v = x[(size_t)r * D + col] + b;
        s += v; s2 += v * v;
    }
    atomicAdd(csum + col, s);
    atomicAdd(csumsq + col, s2);
}

__global__ void bn_params_kernel(const float* __restrict__ csum, const float* __restrict__ csumsq,
                                 const float* __restrict__ g, const float* __restrict__ be,
                                 float* __restrict__ scale, float* __restrict__ shift,
                                 int Ntot, int D) {
    int c = blockIdx.x * blockDim.x + threadIdx.x;
    if (c >= D) return;
    float inv_n = 1.f / (float)Ntot;
    float mu  = csum[c] * inv_n;
    float var = csumsq[c] * inv_n - mu * mu;
    float sc  = g[c] * rsqrtf(var + 1e-5f);
    scale[c] = sc;
    shift[c] = be[c] - mu * sc;
}

// fused (x + bias) -> BN -> ReLU -> f16  (feeds next WMMA GEMM)
__global__ void bn_relu_to_f16_kernel(const float* __restrict__ x, const float* __restrict__ bias,
                                      const float* __restrict__ scale, const float* __restrict__ shift,
                                      _Float16* __restrict__ y, size_t n, int D) {
    size_t i = (size_t)blockIdx.x * blockDim.x + threadIdx.x;
    if (i >= n) return;
    int c = (int)(i % D);
    float v = (x[i] + bias[c]) * scale[c] + shift[c];
    y[i] = (_Float16)(v > 0.f ? v : 0.f);
}

__global__ void bn_relu_to_f32_kernel(const float* __restrict__ x, const float* __restrict__ bias,
                                      const float* __restrict__ scale, const float* __restrict__ shift,
                                      float* __restrict__ y, size_t n, int D) {
    size_t i = (size_t)blockIdx.x * blockDim.x + threadIdx.x;
    if (i >= n) return;
    int c = (int)(i % D);
    float v = (x[i] + bias[c]) * scale[c] + shift[c];
    y[i] = v > 0.f ? v : 0.f;
}

// ---------------- host launcher -------------------------------------------
extern "C" void kernel_launch(void* const* d_in, const int* in_sizes, int n_in,
                              void* d_out, int out_size, void* d_ws, size_t ws_size,
                              hipStream_t stream) {
    const float* x    = (const float*)d_in[0];
    const int*   ei   = (const int*)d_in[1];
    const float* W1   = (const float*)d_in[2];
    const float* atS1 = (const float*)d_in[3];
    const float* atD1 = (const float*)d_in[4];
    const float* b1   = (const float*)d_in[5];
    const float* g1   = (const float*)d_in[6];
    const float* be1  = (const float*)d_in[7];
    const float* W2   = (const float*)d_in[8];
    const float* atS2 = (const float*)d_in[9];
    const float* atD2 = (const float*)d_in[10];
    const float* b2   = (const float*)d_in[11];
    const float* g2   = (const float*)d_in[12];
    const float* be2  = (const float*)d_in[13];

    const int N = in_sizes[0] / IN_DIM;   // 50000
    const int E = in_sizes[1] / 2;        // 800000
    const int* src = ei;
    const int* dst = ei + E;

    // ---- workspace carve-out (256B aligned) ----
    char* ws = (char*)d_ws;
    size_t off = 0;
    auto alloc = [&](size_t bytes) -> char* {
        off = (off + 255) & ~(size_t)255;
        char* p = ws + off;
        off += bytes;
        return p;
    };
    _Float16* X16  = (_Float16*)alloc((size_t)N * IN_DIM * 2);     // reused as A2 f16
    _Float16* W1t  = (_Float16*)alloc((size_t)D1 * IN_DIM * 2);
    _Float16* W2t  = (_Float16*)alloc((size_t)OUT_DIM * D1 * 2);
    float*    h1   = (float*)alloc((size_t)N * D1 * 4);            // reused as h2
    float*    agg1 = (float*)alloc((size_t)N * D1 * 4);            // reused as agg2
    float*    as1  = (float*)alloc((size_t)N * HEADS * 4);
    float*    ad1  = (float*)alloc((size_t)N * HEADS * 4);
    unsigned* m1   = (unsigned*)alloc((size_t)N * HEADS * 4);
    float*    den1 = (float*)alloc((size_t)N * HEADS * 4);
    float*    wbuf = (float*)alloc((size_t)E * HEADS * 4);         // layer2 uses E*1
    float*    as2  = (float*)alloc((size_t)N * 4);
    float*    ad2  = (float*)alloc((size_t)N * 4);
    unsigned* m2   = (unsigned*)alloc((size_t)N * 4);
    float*    den2 = (float*)alloc((size_t)N * 4);
    float*    csum   = (float*)alloc(256 * 4);
    float*    csumsq = (float*)alloc(256 * 4);
    float*    scale  = (float*)alloc(256 * 4);
    float*    shift  = (float*)alloc(256 * 4);
    _Float16* A2   = X16;
    float*    h2   = h1;
    float*    agg2 = agg1;

    // ---- prep: f16 conversions / weight transposes ----
    {
        size_t n = (size_t)N * IN_DIM;
        f32_to_f16_kernel<<<(unsigned)CDIV(n, 256), 256, 0, stream>>>(x, X16, n);
    }
    transpose_to_f16_kernel<<<CDIV(IN_DIM * D1, 256), 256, 0, stream>>>(W1, W1t, IN_DIM, D1);
    transpose_to_f16_kernel<<<CDIV(D1 * OUT_DIM, 256), 256, 0, stream>>>(W2, W2t, D1, OUT_DIM);

    // ================= layer 1 =================
    {   // GEMM1: h1 = X16 x W1 (K=256, Nout=256)
        int tilesM = N / 16, tilesN = D1 / 16;
        int waves = tilesM * tilesN;
        wmma_gemm_f16_kernel<IN_DIM><<<CDIV(waves, 4), 128, 0, stream>>>(
            X16, W1t, h1, tilesM, tilesN, D1);
    }
    att_dots_kernel<<<CDIV(N * HEADS, 4), 128, 0, stream>>>(h1, atS1, atD1, as1, ad1, N, HEADS, HID);

    fill_u32_kernel<<<CDIV((size_t)N * HEADS, 256), 256, 0, stream>>>(m1, ENC_NEGINF, (size_t)N * HEADS);
    fill_u32_kernel<<<CDIV((size_t)N * HEADS, 256), 256, 0, stream>>>((unsigned*)den1, 0u, (size_t)N * HEADS);
    fill_u32_kernel<<<(unsigned)CDIV((size_t)N * D1, 256), 256, 0, stream>>>((unsigned*)agg1, 0u, (size_t)N * D1);

    edge_max_kernel<<<CDIV(E * HEADS, 256), 256, 0, stream>>>(src, dst, as1, ad1, m1, E, HEADS);
    decode_max_kernel<<<CDIV(N * HEADS, 256), 256, 0, stream>>>(m1, N * HEADS);
    edge_w_kernel<<<CDIV(E * HEADS, 256), 256, 0, stream>>>(src, dst, as1, ad1, (const float*)m1, den1, wbuf, E, HEADS);
    edge_aggregate_kernel<<<CDIV(E, 8), 256, 0, stream>>>(src, dst, h1, wbuf, den1, agg1, E, HEADS, HID);

    fill_u32_kernel<<<2, 256, 0, stream>>>((unsigned*)csum, 0u, 256);
    fill_u32_kernel<<<2, 256, 0, stream>>>((unsigned*)csumsq, 0u, 256);
    bn_colstats_kernel<<<200, D1, 0, stream>>>(agg1, b1, csum, csumsq, N, D1, CDIV(N, 200));
    bn_params_kernel<<<1, 256, 0, stream>>>(csum, csumsq, g1, be1, scale, shift, N, D1);
    {
        size_t n = (size_t)N * D1;
        bn_relu_to_f16_kernel<<<(unsigned)CDIV(n, 256), 256, 0, stream>>>(agg1, b1, scale, shift, A2, n, D1);
    }

    // ================= layer 2 =================
    {   // GEMM2: h2 = A2 x W2 (K=256, Nout=128)
        int tilesM = N / 16, tilesN = OUT_DIM / 16;
        int waves = tilesM * tilesN;
        wmma_gemm_f16_kernel<D1><<<CDIV(waves, 4), 128, 0, stream>>>(
            A2, W2t, h2, tilesM, tilesN, OUT_DIM);
    }
    att_dots_kernel<<<CDIV(N, 4), 128, 0, stream>>>(h2, atS2, atD2, as2, ad2, N, 1, OUT_DIM);

    fill_u32_kernel<<<CDIV(N, 256), 256, 0, stream>>>(m2, ENC_NEGINF, (size_t)N);
    fill_u32_kernel<<<CDIV(N, 256), 256, 0, stream>>>((unsigned*)den2, 0u, (size_t)N);
    fill_u32_kernel<<<(unsigned)CDIV((size_t)N * OUT_DIM, 256), 256, 0, stream>>>((unsigned*)agg2, 0u, (size_t)N * OUT_DIM);

    edge_max_kernel<<<CDIV(E, 256), 256, 0, stream>>>(src, dst, as2, ad2, m2, E, 1);
    decode_max_kernel<<<CDIV(N, 256), 256, 0, stream>>>(m2, N);
    edge_w_kernel<<<CDIV(E, 256), 256, 0, stream>>>(src, dst, as2, ad2, (const float*)m2, den2, wbuf, E, 1);
    edge_aggregate_kernel<<<CDIV(E, 8), 256, 0, stream>>>(src, dst, h2, wbuf, den2, agg2, E, 1, OUT_DIM);

    fill_u32_kernel<<<2, 256, 0, stream>>>((unsigned*)csum, 0u, 256);
    fill_u32_kernel<<<2, 256, 0, stream>>>((unsigned*)csumsq, 0u, 256);
    bn_colstats_kernel<<<200, OUT_DIM, 0, stream>>>(agg2, b2, csum, csumsq, N, OUT_DIM, CDIV(N, 200));
    bn_params_kernel<<<1, 256, 0, stream>>>(csum, csumsq, g2, be2, scale, shift, N, OUT_DIM);
    {
        size_t n = (size_t)N * OUT_DIM;
        bn_relu_to_f32_kernel<<<(unsigned)CDIV(n, 256), 256, 0, stream>>>(agg2, b2, scale, shift, (float*)d_out, n, OUT_DIM);
    }
}
